// eFedGauss_78812649881910
// MI455X (gfx1250) — compile-verified, matrix operand order checked by reading, not run on hardware
//
#include <hip/hip_runtime.h>
#include <hip/hip_bf16.h>

// Problem constants from the reference
#define Bn   1024   // batch
#define Cn   512    // clusters
#define Dn   256    // feature dim
#define Kn   10     // classes
#define ROWS 128    // batch rows per block (8 waves x 16 rows)
#define EPSF 1e-12f

// LDS layout (dynamic, 320 KB == WGP max)
#define SH_OFF    0         // S_c as f16 [256][256]           = 131072 B
#define DH_OFF    131072    // diff f16 [128][256]             =  65536 B
#define STAGE_OFF 196608    // 2 x 64KB f32 TDM staging, later reused as
                            // DhT f32 [256][128]              = 131072 B
#define SMEM_TOTAL 327680

typedef __attribute__((ext_vector_type(16))) _Float16 v16h;
typedef __attribute__((ext_vector_type(8)))  _Float16 v8h;
typedef __attribute__((ext_vector_type(4)))  _Float16 v4h;
typedef __attribute__((ext_vector_type(8)))  float    v8f;
typedef __attribute__((ext_vector_type(4)))  float    v4f;
typedef __attribute__((ext_vector_type(4)))  unsigned int u32x4;
typedef __attribute__((ext_vector_type(8)))  int      i32x8;
typedef __attribute__((ext_vector_type(4)))  int      i32x4;

union H16 { v16h h16; v8h h8[2]; };

// ---------------------------------------------------------------------------
// TDM: 1-D async DMA of `nelem` f32 from global -> LDS offset, per CDNA5 D#.
// Group0: count=1 | lds_addr | global_addr[56:0] | type=2 ("image").
// Group1: data_size=2 (4B), tensor_dim0 = tile_dim0 = nelem (<= 65535),
//         tensor_dim1 = 1, tile_dim1/2 = 0 (unused), stride0 = nelem.
// This toolchain exposes the 6-arg builtin (clang-23 / therock headers):
//   (u32x4 g0, i32x8 g1, i32x4 g2, i32x4 g3, i32x8 g4, i32 cpol)
// Tracked by TENSORcnt.
// ---------------------------------------------------------------------------
__device__ __forceinline__ void tdm_load_f32_1d(const float* gsrc,
                                                unsigned lds_off,
                                                unsigned nelem) {
  const unsigned long long ga = (unsigned long long)(const void*)gsrc;
  u32x4 g0;
  g0.x = 1u;                                              // count=1, user D#
  g0.y = lds_off;                                         // lds_addr (bytes)
  g0.z = (unsigned)(ga & 0xFFFFFFFFu);                    // global_addr[31:0]
  g0.w = (unsigned)((ga >> 32) & 0x1FFFFFFu) | (2u << 30);// addr[56:32]|type=2
  i32x8 g1;
  g1[0] = (int)(2u << 16);                  // wg_mask=0, data_size=2 (4 bytes)
  g1[1] = (int)((nelem & 0xFFFFu) << 16);   // tensor_dim0[15:0]
  g1[2] = (int)(((nelem >> 16) & 0xFFFFu) | (1u << 16)); // dim0[31:16],dim1=1
  g1[3] = (int)((nelem & 0xFFFFu) << 16);   // tensor_dim1[31:16]=0, tile_dim0
  g1[4] = 0;                                // tile_dim1 = tile_dim2 = 0
  g1[5] = (int)nelem;                       // tensor_dim0_stride[31:0]
  g1[6] = 0;
  g1[7] = 0;
  i32x4 gz4 = {0, 0, 0, 0};
  i32x8 gz8 = {0, 0, 0, 0, 0, 0, 0, 0};
  __builtin_amdgcn_tensor_load_to_lds(g0, g1, gz4, gz4, gz8, 0);
}

// ---------------------------------------------------------------------------
// Kernel 1: d2[b,c] = (x_b - mu_c)^T S_c (x_b - mu_c), via WMMA f16->f32.
// S_c symmetric, so the B-operand column equals a row of S_c (no transpose).
// Grid: (Cn, Bn/ROWS). Block: 256 threads (8 wave32).
// ---------------------------------------------------------------------------
__global__ void __launch_bounds__(256)
mahalanobis_d2_kernel(const float* __restrict__ X,
                      const float* __restrict__ Mu,
                      const float* __restrict__ S,
                      float* __restrict__ d2g) {
  extern __shared__ char smem[];
  _Float16* Sh   = (_Float16*)(smem + SH_OFF);    // [Dn][Dn] f16
  _Float16* Dh   = (_Float16*)(smem + DH_OFF);    // [ROWS][Dn] f16
  float*    DhT  = (float*)(smem + STAGE_OFF);    // [Dn][ROWS] f32 (late)

  const int c    = blockIdx.x;
  const int row0 = blockIdx.y * ROWS;
  const int tid  = threadIdx.x;
  const int lane = tid & 31;
  const int wave = tid >> 5;
  const int mrow = lane & 15;   // A-layout M / C-D-layout N  (lane % 16)
  const int grp  = lane >> 4;   // lane half-group

  const float* Sc = S + (size_t)c * (Dn * Dn);

  // --- Kick off async S_c DMA: chunks 0,1 (64 KB each) into staging bufs ---
  if (wave == 0) {
    tdm_load_f32_1d(Sc,          STAGE_OFF,         16384);
    tdm_load_f32_1d(Sc + 16384,  STAGE_OFF + 65536, 16384);
  }

  // --- Overlap with DMA: diff = X_tile - mu_c -> f16 LDS ---
  const float* mu = Mu + (size_t)c * Dn;
  for (int i = tid * 4; i < ROWS * Dn; i += 256 * 4) {
    int m = i >> 8;      // / Dn
    int k = i & (Dn - 1);
    v4f x4 = *(const v4f*)(X + (size_t)(row0 + m) * Dn + k);
    v4f u4 = *(const v4f*)(mu + k);
    v4h h;
    h.x = (_Float16)(x4.x - u4.x); h.y = (_Float16)(x4.y - u4.y);
    h.z = (_Float16)(x4.z - u4.z); h.w = (_Float16)(x4.w - u4.w);
    *(v4h*)(Dh + i) = h;
  }

  // --- Double-buffered pipeline: drain chunk, convert f32->f16, refill ---
#pragma unroll
  for (int ch = 0; ch < 4; ++ch) {
    if (wave == 0) {
      if (ch == 3) __builtin_amdgcn_s_wait_tensorcnt(0);
      else         __builtin_amdgcn_s_wait_tensorcnt(1);
    }
    __syncthreads();   // chunk `ch` now visible in staging buf (ch & 1)
    const float* buf = (const float*)(smem + STAGE_OFF + (ch & 1) * 65536);
    _Float16* dst = Sh + ch * 16384;
    for (int i = tid * 4; i < 16384; i += 256 * 4) {
      v4f s4 = *(const v4f*)(buf + i);
      v4h h;
      h.x = (_Float16)s4.x; h.y = (_Float16)s4.y;
      h.z = (_Float16)s4.z; h.w = (_Float16)s4.w;
      *(v4h*)(dst + i) = h;
    }
    __syncthreads();   // everyone done reading buf before DMA overwrites it
    if (wave == 0 && ch + 2 < 4)
      tdm_load_f32_1d(Sc + (ch + 2) * 16384, STAGE_OFF + (ch & 1) * 65536, 16384);
  }

  // --- Staging now free: build transposed f32 diff for the dot epilogue ---
  for (int i = tid * 4; i < ROWS * Dn; i += 256 * 4) {
    int m = i >> 8;
    int k = i & (Dn - 1);
    v4h d4 = *(const v4h*)(Dh + i);
    DhT[(k + 0) * ROWS + m] = (float)d4.x;
    DhT[(k + 1) * ROWS + m] = (float)d4.y;
    DhT[(k + 2) * ROWS + m] = (float)d4.z;
    DhT[(k + 3) * ROWS + m] = (float)d4.w;
  }
  __syncthreads();

  // --- Per-wave A operand (16 rows x 256 K), ISA 16-bit A layout:
  //     lane group 0 holds K {kc*32+0..7, kc*32+16..23}, group 1 the rest.
  const int arow = wave * 16 + mrow;
  H16 a[8];
#pragma unroll
  for (int kc = 0; kc < 8; ++kc) {
    const int ks = kc * 32 + grp * 8;
    a[kc].h8[0] = *(const v8h*)(Dh + arow * Dn + ks);
    a[kc].h8[1] = *(const v8h*)(Dh + arow * Dn + ks + 16);
  }

  // --- N sweep: Y_tile = Diff x S via 8 WMMAs, fused row-dot epilogue ---
  float acc[8] = {0.f, 0.f, 0.f, 0.f, 0.f, 0.f, 0.f, 0.f};
  for (int nt = 0; nt < 16; ++nt) {
    v8f cacc = {};
    const int ncol = nt * 16 + mrow;   // B column / C-D N for this lane
#pragma unroll
    for (int kc = 0; kc < 8; ++kc) {
      const int ks = kc * 32 + grp * 8;
      H16 b;
      // S symmetric: column ncol of S == row ncol of S (row-major LDS)
      b.h8[0] = *(const v8h*)(Sh + ncol * Dn + ks);
      b.h8[1] = *(const v8h*)(Sh + ncol * Dn + ks + 16);
      cacc = __builtin_amdgcn_wmma_f32_16x16x32_f16(
          false, a[kc].h16, false, b.h16, (short)0, cacc, false, false);
    }
    // C/D layout: VGPR v, lane l -> (m = v + 8*grp, n = l%16).
    // DhT gives the 8 diff values contiguously: [ncol][wave*16 + grp*8 + v].
    const float* dptr = DhT + ncol * ROWS + wave * 16 + grp * 8;
    v4f t0 = *(const v4f*)(dptr);
    v4f t1 = *(const v4f*)(dptr + 4);
    acc[0] += cacc[0] * t0.x; acc[1] += cacc[1] * t0.y;
    acc[2] += cacc[2] * t0.z; acc[3] += cacc[3] * t0.w;
    acc[4] += cacc[4] * t1.x; acc[5] += cacc[5] * t1.y;
    acc[6] += cacc[6] * t1.z; acc[7] += cacc[7] * t1.w;
  }

  // --- Reduce partial dots across the 16 lanes sharing each row ---
#pragma unroll
  for (int v = 0; v < 8; ++v) {
#pragma unroll
    for (int off = 1; off < 16; off <<= 1)
      acc[v] += __shfl_xor(acc[v], off, 32);
  }
  if (mrow == 0) {   // lanes 0 (rows 0..7) and 16 (rows 8..15)
#pragma unroll
    for (int v = 0; v < 8; ++v) {
      const int m = v + grp * 8;
      d2g[(size_t)(row0 + wave * 16 + m) * Cn + c] = acc[v];
    }
  }
}

// ---------------------------------------------------------------------------
// Kernel 2: Gamma = exp(-0.5 d2); label_scores = (Gamma @ dist)/(sum+eps);
// preds = argmax scores; clusters = argmax Gamma == argmin d2.
// One wave32 per batch row. Grid: Bn/8 blocks of 256 threads.
// ---------------------------------------------------------------------------
__global__ void __launch_bounds__(256)
defuzz_kernel(const float* __restrict__ d2g,
              const float* __restrict__ cld,
              float* __restrict__ out) {
  __shared__ float cldS[Cn * Kn];   // 20 KB
  const int tid = threadIdx.x;
  for (int i = tid; i < Cn * Kn; i += 256) cldS[i] = cld[i];
  __syncthreads();

  const int lane = tid & 31;
  const int wave = tid >> 5;
  const int b = blockIdx.x * 8 + wave;

  float sum = 0.f;
  float dmin = 3.4e38f;
  int   cbest = 0;
  float sc[Kn];
#pragma unroll
  for (int k = 0; k < Kn; ++k) sc[k] = 0.f;

  for (int ci = 0; ci < Cn / 32; ++ci) {
    const int cc = ci * 32 + lane;
    const float d = d2g[(size_t)b * Cn + cc];
    const float g = expf(-0.5f * d);
    sum += g;
    if (d < dmin) { dmin = d; cbest = cc; }
#pragma unroll
    for (int k = 0; k < Kn; ++k) sc[k] += g * cldS[cc * Kn + k];
  }

  // wave32 reductions
#pragma unroll
  for (int off = 1; off < 32; off <<= 1) {
    sum += __shfl_xor(sum, off, 32);
    const float od = __shfl_xor(dmin, off, 32);
    const int   oc = __shfl_xor(cbest, off, 32);
    if (od < dmin || (od == dmin && oc < cbest)) { dmin = od; cbest = oc; }
#pragma unroll
    for (int k = 0; k < Kn; ++k) sc[k] += __shfl_xor(sc[k], off, 32);
  }

  if (lane == 0) {
    const float inv = 1.f / (sum + EPSF);
    float best = -3.4e38f;
    int bk = 0;
#pragma unroll
    for (int k = 0; k < Kn; ++k) {
      const float s = sc[k] * inv;
      out[(size_t)b * Kn + k] = s;
      if (s > best) { best = s; bk = k; }
    }
    out[Bn * Kn + b]      = (float)bk;     // preds_max
    out[Bn * Kn + Bn + b] = (float)cbest;  // clusters
  }
}

// ---------------------------------------------------------------------------
extern "C" void kernel_launch(void* const* d_in, const int* in_sizes, int n_in,
                              void* d_out, int out_size, void* d_ws, size_t ws_size,
                              hipStream_t stream) {
  const float* X   = (const float*)d_in[0];   // [B, D]
  const float* Mu  = (const float*)d_in[1];   // [C, D]
  const float* S   = (const float*)d_in[2];   // [C, D, D]
  const float* cld = (const float*)d_in[3];   // [C, K]
  float* out = (float*)d_out;
  float* d2g = (float*)d_ws;                  // [B, C] f32 scratch (2 MB)

  dim3 grid1(Cn, Bn / ROWS);
  mahalanobis_d2_kernel<<<grid1, 256, SMEM_TOTAL, stream>>>(X, Mu, S, d2g);

  defuzz_kernel<<<Bn / 8, 256, 0, stream>>>(d2g, cld, out);
}
